// E26DualMemoryElmanCell_66202625901139
// MI455X (gfx1250) — compile-verified, matrix operand order checked by reading, not run on hardware
//
#include <hip/hip_runtime.h>

// ---- problem constants -------------------------------------------------
#define BB   16
#define TT   1024
#define DD   1024
#define NN   8
#define SCALE 0.03125f          // 1/sqrt(1024)

// ---- scan-kernel tiling ------------------------------------------------
#define NWG  32                 // persistent workgroups, each owns D/NWG cols
#define COLS 32                 // columns per workgroup
#define SCAN_THREADS 128        // 4 waves (wave32)

typedef __attribute__((ext_vector_type(16))) __bf16 v16bf;
typedef __attribute__((ext_vector_type(8)))  float  v8f;
typedef __attribute__((ext_vector_type(4)))  unsigned v4u;
typedef __attribute__((ext_vector_type(8)))  unsigned v8u;

// ---- fp32 -> bf16 conversion ------------------------------------------
__global__ void cvt_f32_bf16(const float* __restrict__ s, __bf16* __restrict__ d, int n) {
    int i = blockIdx.x * blockDim.x + threadIdx.x;
    int stride = gridDim.x * blockDim.x;
    for (; i < n; i += stride) d[i] = (__bf16)s[i];
}

// ---- big GEMM: x_proj[bt, e] = sum_d x[bt, d] * W_x[e, d] --------------
// each wave computes a 16x64 strip (4 C tiles) sharing one A fragment.
__global__ __launch_bounds__(256) void xproj_kernel(
    const __bf16* __restrict__ xb,      // [16384, 1024]
    const __bf16* __restrict__ wxb,     // [1024, 1024] (row e, col d)
    float* __restrict__ xproj)          // [16384, 1024]
{
    const int lane = threadIdx.x & 31;
    const int wave = threadIdx.x >> 5;
    const int job  = blockIdx.x * 8 + wave;       // 16384 wave-jobs
    const int mt   = job >> 4;                    // 1024 row-tiles
    const int nt0  = (job & 15) << 2;             // base of 4 col-tiles
    const int m    = lane & 15;
    const int ks   = (lane >> 4) << 4;            // 0 or 16

    const __bf16* arow = xb  + (size_t)(mt * 16 + m) * DD;
    const __bf16* brow = wxb + (size_t)(nt0 * 16 + m) * DD;  // lane%16 -> N col
    v8f a0 = {}, a1 = {}, a2 = {}, a3 = {};
    for (int kt = 0; kt < 32; ++kt) {
        const int kb = kt * 32 + ks;
        v16bf a  = *(const v16bf*)(arow + kb);
        v16bf b0 = *(const v16bf*)(brow + kb);
        v16bf b1 = *(const v16bf*)(brow + 16 * DD + kb);
        v16bf b2 = *(const v16bf*)(brow + 32 * DD + kb);
        v16bf b3 = *(const v16bf*)(brow + 48 * DD + kb);
        a0 = __builtin_amdgcn_wmma_f32_16x16x32_bf16(false, a, false, b0, (short)0, a0, false, false);
        a1 = __builtin_amdgcn_wmma_f32_16x16x32_bf16(false, a, false, b1, (short)0, a1, false, false);
        a2 = __builtin_amdgcn_wmma_f32_16x16x32_bf16(false, a, false, b2, (short)0, a2, false, false);
        a3 = __builtin_amdgcn_wmma_f32_16x16x32_bf16(false, a, false, b3, (short)0, a3, false, false);
    }
    const int row0 = mt * 16 + 8 * (lane >> 4);   // C layout: VGPR r -> M=r(+8)
    const int col  = (lane & 15);
#pragma unroll
    for (int r = 0; r < 8; ++r) {
        float* orow = xproj + (size_t)(row0 + r) * DD + nt0 * 16 + col;
        orow[0]  = a0[r];
        orow[16] = a1[r];
        orow[32] = a2[r];
        orow[48] = a3[r];
    }
}

// ---- TDM: DMA a 2-D bf16 tile (rows x cols, row-major, stride==cols) ----
// from global memory into LDS.  D# layout per CDNA5 ISA 08_async_tensor §8.
__device__ __forceinline__ void tdm_load_2d_bf16(unsigned lds_off,
                                                 const void* gptr,
                                                 unsigned rows, unsigned cols) {
    const unsigned long long ga = (unsigned long long)gptr;
    v4u g0;
    g0[0] = 1u;                                   // count=1, is_restore=0
    g0[1] = lds_off;                              // lds_addr
    g0[2] = (unsigned)ga;                         // global_addr[31:0]
    g0[3] = ((unsigned)(ga >> 32) & 0x01FFFFFFu)  // global_addr[56:32]
            | (2u << 30);                         // type=2 ("image")
    v8u g1;
    g1[0] = 1u << 16;                             // data_size=1 (2 bytes)
    g1[1] = (cols & 0xFFFFu) << 16;               // tensor_dim0[15:0] @bit48
    g1[2] = ((cols >> 16) & 0xFFFFu)              // tensor_dim0[31:16]
            | ((rows & 0xFFFFu) << 16);           // tensor_dim1[15:0]
    g1[3] = ((rows >> 16) & 0xFFFFu)              // tensor_dim1[31:16]
            | ((cols & 0xFFFFu) << 16);           // tile_dim0
    g1[4] = rows & 0xFFFFu;                       // tile_dim1 (tile_dim2=0)
    g1[5] = cols;                                 // tensor_dim0_stride[31:0]
    g1[6] = ((cols >> 32) & 0xFFFFu)              // dim0_stride[47:32] (=0)
            | ((cols & 0xFFFFu) << 16);           // tensor_dim1_stride[15:0]
    g1[7] = (cols >> 16);                         // tensor_dim1_stride[47:16]
    asm volatile("tensor_load_to_lds %0, %1" :: "s"(g0), "s"(g1) : "memory");
}

// ---- helpers for the persistent scan kernel ----------------------------
__device__ __forceinline__ void grid_barrier(int* bar, int id) {
    __syncthreads();
    __threadfence();
    if (threadIdx.x == 0) {
        atomicAdd(&bar[id], 1);
        while (__hip_atomic_load(&bar[id], __ATOMIC_RELAXED,
                                 __HIP_MEMORY_SCOPE_AGENT) < NWG) {
            __builtin_amdgcn_s_sleep(1);
        }
    }
    __syncthreads();
    __threadfence();
}

// M=16 x N=32 x K=1024 GEMM: A in global (bf16 [16,1024]), B slice in LDS
// ([32 local cols][1024 k], bf16). 4 waves: (ntile, khalf); K-halves reduced
// through LDS. Result -> Cout [16][COLS] fp32 in LDS.
__device__ __forceinline__ void gemm_tile(const __bf16* __restrict__ A,
                                          const __bf16* __restrict__ Blds,
                                          float* __restrict__ Cout,
                                          float* __restrict__ sRed,
                                          int wave, int lane)
{
    const int ntile = wave & 1;
    const int khalf = wave >> 1;
    const int m  = lane & 15;
    const int ks = (lane >> 4) << 4;
    v8f acc = {};
    for (int kt = 0; kt < 16; ++kt) {
        const int kb = (khalf * 16 + kt) * 32 + ks;
        v16bf a = *(const v16bf*)(A + m * DD + kb);
        v16bf b = *(const v16bf*)(Blds + (ntile * 16 + m) * DD + kb);  // lane%16 -> n
        acc = __builtin_amdgcn_wmma_f32_16x16x32_bf16(false, a, false, b,
                                                      (short)0, acc, false, false);
    }
    if (khalf == 1) {
#pragma unroll
        for (int r = 0; r < 8; ++r) sRed[ntile * 256 + r * 32 + lane] = acc[r];
    }
    __syncthreads();
    if (khalf == 0) {
#pragma unroll
        for (int r = 0; r < 8; ++r) {
            float v  = acc[r] + sRed[ntile * 256 + r * 32 + lane];
            int   mm = r + 8 * (lane >> 4);
            int   nn = ntile * 16 + (lane & 15);
            Cout[mm * COLS + nn] = v;
        }
    }
    __syncthreads();
}

__device__ __forceinline__ void softmax8(const float* __restrict__ g,
                                         float* __restrict__ attn, int tid) {
    // tid < 128 always; thread (b,n) computes its own attention weight.
    const int b = tid >> 3, n = tid & 7;
    float s[NN];
    float mx = -1e30f;
#pragma unroll
    for (int k = 0; k < NN; ++k) { s[k] = g[b * NN + k] * SCALE; mx = fmaxf(mx, s[k]); }
    float den = 0.f;
#pragma unroll
    for (int k = 0; k < NN; ++k) den += __expf(s[k] - mx);
    attn[b * NN + n] = __expf(s[n] - mx) / den;
    __syncthreads();
}

// ---- persistent recurrent scan kernel ----------------------------------
__global__ __launch_bounds__(SCAN_THREADS) void scan_kernel(
    const float*  __restrict__ xproj,   // [16, 1024, 1024] fp32
    const __bf16* __restrict__ whb,     // [1024, 1024] bf16
    const __bf16* __restrict__ wwb,     // [1024, 1024] bf16
    const float*  __restrict__ bh,      // [1024] fp32
    __bf16*       __restrict__ hbuf,    // 2 x [16, 1024] bf16 (double buffer, zeroed)
    float*        __restrict__ g_rd,    // [1024, 128] (zeroed)
    float*        __restrict__ g_wr,    // [1024, 128] (zeroed)
    int*          __restrict__ g_bar,   // [3072] (zeroed)
    float*        __restrict__ out_all, // [16, 1024, 1024]
    float*        __restrict__ out_tape,// [16, 8, 1024]
    float*        __restrict__ out_last)// [16, 1024]
{
    __shared__ __attribute__((aligned(32))) __bf16 sWh[COLS * DD];   // 64 KB
    __shared__ __attribute__((aligned(32))) __bf16 sWw[COLS * DD];   // 64 KB
    __shared__ float sTape[BB * NN * COLS];                          // 16 KB
    __shared__ float sC[BB * COLS];                                  // 2 KB
    __shared__ float sRed[2 * 256];                                  // 2 KB
    __shared__ float sAttn[BB * NN];                                 // 0.5 KB

    const int tid  = threadIdx.x;
    const int wg   = blockIdx.x;
    const int c0   = wg * COLS;
    const int lane = tid & 31;
    const int wave = tid >> 5;

    // ---- preload this WG's weight slices into LDS via the Tensor Data
    // Mover (resident for all T steps).  Wave 0 issues both DMAs.
    if (tid < 32) {
        const unsigned lds_wh = (unsigned)(unsigned long long)(const void*)sWh;
        const unsigned lds_ww = (unsigned)(unsigned long long)(const void*)sWw;
        tdm_load_2d_bf16(lds_wh, whb + (size_t)c0 * DD, COLS, DD);
        tdm_load_2d_bf16(lds_ww, wwb + (size_t)c0 * DD, COLS, DD);
        __builtin_amdgcn_s_wait_tensorcnt(0);
    }
    for (int i = tid; i < BB * NN * COLS; i += SCAN_THREADS) sTape[i] = 0.f;
    __syncthreads();

    const int sb = tid >> 3, sn = tid & 7;   // (batch, slot) this thread scores

    for (int t = 0; t < TT; ++t) {
        const __bf16* h_cur = hbuf + ((t & 1) ? BB * DD : 0);
        __bf16*       h_nxt = hbuf + ((t & 1) ? 0 : BB * DD);

        // --- read-score partials over our column slice ------------------
        {
            float p = 0.f;
            for (int j = 0; j < COLS; ++j)
                p += (float)h_cur[sb * DD + c0 + j] * sTape[(sb * NN + sn) * COLS + j];
            atomicAdd(&g_rd[(size_t)t * 128 + tid], p);
        }
        // --- prefetch next step's x_proj slice (the only cold stream) ---
        if (t + 1 < TT) {
#pragma unroll
            for (int jj = 0; jj < 4; ++jj) {
                const int flat = tid + SCAN_THREADS * jj;
                const int b = flat >> 5, j = flat & 31;
                __builtin_prefetch(&xproj[((size_t)b * TT + t + 1) * DD + c0 + j], 0, 1);
            }
        }
        // --- Rh = h_work @ W_h.T  (our 32 cols) -> sC -------------------
        gemm_tile(h_cur, sWh, sC, sRed, wave, lane);

        grid_barrier(g_bar, t * 3 + 0);
        softmax8(&g_rd[(size_t)t * 128], sAttn, tid);

        // --- h_new = tanh(x_t + Rh + read_val + b_h) --------------------
#pragma unroll
        for (int jj = 0; jj < 4; ++jj) {
            const int flat = tid + SCAN_THREADS * jj;   // 512 (b, j) pairs
            const int b = flat >> 5, j = flat & 31;
            float rv = 0.f;
#pragma unroll
            for (int n = 0; n < NN; ++n)
                rv += sAttn[b * NN + n] * sTape[(b * NN + n) * COLS + j];
            const float hv = tanhf(xproj[((size_t)b * TT + t) * DD + c0 + j]
                                   + sC[b * COLS + j] + rv + bh[c0 + j]);
            h_nxt[b * DD + c0 + j] = (__bf16)hv;
            out_all[((size_t)b * TT + t) * DD + c0 + j] = hv;
        }

        grid_barrier(g_bar, t * 3 + 1);   // full h_new visible device-wide

        // --- write_val = h_new @ W_write.T (our 32 cols) -> sC ----------
        gemm_tile(h_nxt, sWw, sC, sRed, wave, lane);

        // --- write-score partials ---------------------------------------
        {
            float p = 0.f;
            for (int j = 0; j < COLS; ++j)
                p += sC[sb * COLS + j] * sTape[(sb * NN + sn) * COLS + j];
            atomicAdd(&g_wr[(size_t)t * 128 + tid], p);
        }
        grid_barrier(g_bar, t * 3 + 2);
        softmax8(&g_wr[(size_t)t * 128], sAttn, tid);

        // --- tape update -------------------------------------------------
#pragma unroll
        for (int jj = 0; jj < 4; ++jj) {
            const int flat = tid + SCAN_THREADS * jj;
            const int b = flat >> 5, j = flat & 31;
            const float wv = sC[b * COLS + j];
#pragma unroll
            for (int n = 0; n < NN; ++n) {
                const float a = sAttn[b * NN + n];
                float* p = &sTape[(b * NN + n) * COLS + j];
                *p = *p * (1.f - a) + wv * a;
            }
        }
        __syncthreads();
    }

    // --- epilogue: final tape + last hidden state ------------------------
    for (int idx = tid; idx < BB * NN * COLS; idx += SCAN_THREADS) {
        const int b = idx / (NN * COLS);
        const int r = idx % (NN * COLS);
        const int n = r / COLS, j = r % COLS;
        out_tape[((size_t)b * NN + n) * DD + c0 + j] = sTape[(b * NN + n) * COLS + j];
    }
    __threadfence();
    __syncthreads();
    for (int idx = tid; idx < BB * COLS; idx += SCAN_THREADS) {
        const int b = idx / COLS, j = idx % COLS;
        out_last[(size_t)b * DD + c0 + j] =
            out_all[((size_t)b * TT + (TT - 1)) * DD + c0 + j];
    }
}

// ---- host launcher ------------------------------------------------------
extern "C" void kernel_launch(void* const* d_in, const int* in_sizes, int n_in,
                              void* d_out, int out_size, void* d_ws, size_t ws_size,
                              hipStream_t stream) {
    const float* x  = (const float*)d_in[0];   // [16,1024,1024]
    const float* Wh = (const float*)d_in[1];   // [1024,1024]
    const float* Wx = (const float*)d_in[2];   // [1024,1024]
    const float* bh = (const float*)d_in[3];   // [1024]
    const float* Ww = (const float*)d_in[4];   // [1024,1024]

    char* ws = (char*)d_ws;
    size_t off = 0;
    __bf16* xb   = (__bf16*)(ws + off); off += (size_t)BB * TT * DD * 2;   // 32 MB
    __bf16* wxb  = (__bf16*)(ws + off); off += (size_t)DD * DD * 2;        // 2 MB
    __bf16* whb  = (__bf16*)(ws + off); off += (size_t)DD * DD * 2;        // 2 MB
    __bf16* wwb  = (__bf16*)(ws + off); off += (size_t)DD * DD * 2;        // 2 MB
    float*  xproj= (float*)(ws + off);  off += (size_t)BB * TT * DD * 4;   // 64 MB
    char*   z0   = ws + off;
    __bf16* hbuf = (__bf16*)(ws + off); off += (size_t)2 * BB * DD * 2;
    float*  g_rd = (float*)(ws + off);  off += (size_t)TT * 128 * 4;
    float*  g_wr = (float*)(ws + off);  off += (size_t)TT * 128 * 4;
    int*    g_bar= (int*)(ws + off);    off += (size_t)3 * TT * 4;
    hipMemsetAsync(z0, 0, (size_t)((ws + off) - z0), stream);

    cvt_f32_bf16<<<2048, 256, 0, stream>>>(x,  xb,  BB * TT * DD);
    cvt_f32_bf16<<<512,  256, 0, stream>>>(Wx, wxb, DD * DD);
    cvt_f32_bf16<<<512,  256, 0, stream>>>(Wh, whb, DD * DD);
    cvt_f32_bf16<<<512,  256, 0, stream>>>(Ww, wwb, DD * DD);

    xproj_kernel<<<2048, 256, 0, stream>>>(xb, wxb, xproj);

    float* out_all  = (float*)d_out;
    float* out_tape = out_all  + (size_t)BB * TT * DD;
    float* out_last = out_tape + (size_t)BB * NN * DD;
    scan_kernel<<<NWG, SCAN_THREADS, 0, stream>>>(
        xproj, whb, wwb, bh, hbuf, g_rd, g_wr, g_bar,
        out_all, out_tape, out_last);
}
